// CustomLinear_39341900431501
// MI455X (gfx1250) — compile-verified, matrix operand order checked by reading, not run on hardware
//
#include <hip/hip_runtime.h>

// ---------------------------------------------------------------------------
// LoRA linear, MI455X (gfx1250, wave32, WMMA).
//   out[8192,4096] = x[8192,4096] @ (W + 2*B@A)^T
// Strategy:
//   1) fuse_w_bf16 : W' = W + 2*B*A  (rank 16), convert to bf16  -> ws
//   2) cvt_x_bf16  : x f32 -> bf16                               -> ws
//   3) gemm_bf16_wmma : 256x128x64 LDS-tiled NT GEMM, 8 waves/block (4x2),
//      each wave 64x64 output = 32x v_wmma_f32_16x16x32_bf16 per K-step.
//      Double-buffered LDS fed by GLOBAL_LOAD_ASYNC_TO_LDS_B128 (ASYNCcnt);
//      one 128B row-slice per thread so global/LDS share literal offsets.
// Workspace use: 64 MB (x bf16) + 32 MB (W' bf16) = 96 MB of d_ws.
// ---------------------------------------------------------------------------

typedef __attribute__((ext_vector_type(16))) __bf16 bf16x16;
typedef __attribute__((ext_vector_type(8)))  float  f32x8;
typedef int v4i __attribute__((vector_size(16)));   // matches builtin param type

#define M_DIM 8192
#define N_DIM 4096
#define K_DIM 4096
#define R_DIM 16
#define ALPHA 2.0f

#define BM 256
#define BN 128
#define BK 64
#define LDS_STRIDE 72   // 64 bf16 + 8 pad -> conflict-free ds_load_b128 gathers

#if defined(__HIP_DEVICE_COMPILE__) &&                                   \
    __has_builtin(__builtin_amdgcn_global_load_async_to_lds_b128) &&     \
    __has_builtin(__builtin_amdgcn_s_wait_asynccnt)
#define USE_ASYNC_LDS 1
#else
#define USE_ASYNC_LDS 0
#endif

__device__ __forceinline__ unsigned short f2bf(float f) {
    // round-to-nearest-even f32 -> bf16
    unsigned int u = __float_as_uint(f);
    unsigned int r = u + 0x7FFFu + ((u >> 16) & 1u);
    return (unsigned short)(r >> 16);
}

union FragCast {
    uint4   u[2];
    bf16x16 v;
};

// 16-byte global -> LDS copy at byte offset OFF (applied to BOTH addresses,
// per ISA async-load semantics). Async DMA path (ASYNCcnt) or sync fallback.
template <int OFF>
__device__ __forceinline__ void copy16_g2l(const unsigned short* g, unsigned short* l) {
#if USE_ASYNC_LDS
    __builtin_amdgcn_global_load_async_to_lds_b128(
        (__attribute__((address_space(1))) v4i*)g,
        (__attribute__((address_space(3))) v4i*)l,
        /*offset=*/OFF, /*cpol=*/0);
#else
    *(uint4*)(l + OFF / 2) = *(const uint4*)(g + OFF / 2);
#endif
}

__device__ __forceinline__ void wait_async_lds() {
#if USE_ASYNC_LDS
    __builtin_amdgcn_s_wait_asynccnt(0);
#endif
}

// ---- x (f32) -> bf16, 8 elements / thread -------------------------------
__global__ __launch_bounds__(256) void cvt_x_bf16(const float* __restrict__ x,
                                                  unsigned short* __restrict__ xbf) {
    size_t idx = (size_t)blockIdx.x * 256 + threadIdx.x;
    const float4* xv = (const float4*)x;
    float4 a = xv[idx * 2 + 0];
    float4 b = xv[idx * 2 + 1];
    uint4 r;
    r.x = (unsigned)f2bf(a.x) | ((unsigned)f2bf(a.y) << 16);
    r.y = (unsigned)f2bf(a.z) | ((unsigned)f2bf(a.w) << 16);
    r.z = (unsigned)f2bf(b.x) | ((unsigned)f2bf(b.y) << 16);
    r.w = (unsigned)f2bf(b.z) | ((unsigned)f2bf(b.w) << 16);
    ((uint4*)xbf)[idx] = r;
}

// ---- W' = W + 2*B*A (rank 16), -> bf16, 2 elements / thread -------------
__global__ __launch_bounds__(256) void fuse_w_bf16(const float* __restrict__ W,
                                                   const float* __restrict__ lA,
                                                   const float* __restrict__ lB,
                                                   unsigned short* __restrict__ wbf) {
    size_t idx = (size_t)blockIdx.x * 256 + threadIdx.x;   // one per 2 elems
    int o = (int)(idx >> 11);            // / (K_DIM/2)
    int i = ((int)idx & 2047) * 2;
    float s0 = W[(size_t)o * K_DIM + i];
    float s1 = W[(size_t)o * K_DIM + i + 1];
#pragma unroll
    for (int r = 0; r < R_DIM; ++r) {
        float b = ALPHA * lB[o * R_DIM + r];
        s0 += b * lA[r * K_DIM + i];
        s1 += b * lA[r * K_DIM + i + 1];
    }
    ((unsigned int*)wbf)[idx] = (unsigned)f2bf(s0) | ((unsigned)f2bf(s1) << 16);
}

// ---- bf16 NT GEMM with WMMA, async double-buffered LDS ------------------
__global__ __launch_bounds__(256) void gemm_bf16_wmma(const unsigned short* __restrict__ Xb,
                                                      const unsigned short* __restrict__ Wb,
                                                      float* __restrict__ out) {
    __shared__ unsigned short sX[2][BM * LDS_STRIDE];   // 2 x 36 KB
    __shared__ unsigned short sW[2][BN * LDS_STRIDE];   // 2 x 18 KB

    const int tid  = threadIdx.x;
    const int lane = tid & 31;
    const int wave = tid >> 5;
    const int wrow = wave >> 1;      // 0..3 : 64 output rows each
    const int wcol = wave & 1;       // 0..1 : 64 output cols each
    const int m0 = blockIdx.y * BM;
    const int n0 = blockIdx.x * BN;

    // Copy assignment: thread t owns X-tile row t (8x16B chunks) and half of
    // W-tile row t/2 (4x16B chunks) -> global/LDS deltas both 16B (imm OFF).
    const unsigned short* gx = Xb + (size_t)(m0 + tid) * K_DIM;
    const unsigned short* gw = Wb + (size_t)(n0 + (tid >> 1)) * K_DIM + (tid & 1) * 32;
    unsigned short* lx[2] = { &sX[0][tid * LDS_STRIDE],
                              &sX[1][tid * LDS_STRIDE] };
    unsigned short* lw[2] = { &sW[0][(tid >> 1) * LDS_STRIDE + (tid & 1) * 32],
                              &sW[1][(tid >> 1) * LDS_STRIDE + (tid & 1) * 32] };

    // Fragment gather coords (ISA 7.12.2, 16-bit A/B 16x32 layout)
    const int laneM = lane & 15;
    const int kh    = (lane >> 4) << 3;     // 0 or 8

    f32x8 acc[4][4] = {};

    auto issue_tile = [&](int buf) {
        unsigned short* dx = lx[buf];
        unsigned short* dw = lw[buf];
        copy16_g2l<0>(gx, dx);   copy16_g2l<16>(gx, dx);
        copy16_g2l<32>(gx, dx);  copy16_g2l<48>(gx, dx);
        copy16_g2l<64>(gx, dx);  copy16_g2l<80>(gx, dx);
        copy16_g2l<96>(gx, dx);  copy16_g2l<112>(gx, dx);
        copy16_g2l<0>(gw, dw);   copy16_g2l<16>(gw, dw);
        copy16_g2l<32>(gw, dw);  copy16_g2l<48>(gw, dw);
        gx += BK;                // advance one K-tile
        gw += BK;
    };

    const int KT = K_DIM / BK;              // 64 K-steps

    issue_tile(0);
    wait_async_lds();
    __syncthreads();

    for (int kt = 0; kt < KT; ++kt) {
        const int buf = kt & 1;

        // start DMA of next tile into the other buffer (reads of it finished
        // at the barrier that ended iteration kt-1)
        if (kt + 1 < KT) issue_tile(buf ^ 1);

#pragma unroll
        for (int s = 0; s < 2; ++s) {       // two K=32 sub-steps
            const int ko = s * 32 + kh;
            bf16x16 bfr[4];
#pragma unroll
            for (int j = 0; j < 4; ++j) {
                const unsigned short* p =
                    &sW[buf][(wcol * 64 + j * 16 + laneM) * LDS_STRIDE + ko];
                FragCast f;
                f.u[0] = *(const uint4*)(p);
                f.u[1] = *(const uint4*)(p + 16);
                bfr[j] = f.v;
            }
#pragma unroll
            for (int i = 0; i < 4; ++i) {
                const unsigned short* p =
                    &sX[buf][(wrow * 64 + i * 16 + laneM) * LDS_STRIDE + ko];
                FragCast f;
                f.u[0] = *(const uint4*)(p);
                f.u[1] = *(const uint4*)(p + 16);
                bf16x16 afr = f.v;
#pragma unroll
                for (int j = 0; j < 4; ++j)
                    acc[i][j] = __builtin_amdgcn_wmma_f32_16x16x32_bf16(
                        /*neg_a=*/false, afr,
                        /*neg_b=*/false, bfr[j],
                        /*c_mod=*/(short)0, acc[i][j],
                        /*reuse_a=*/false, /*reuse_b=*/false);
            }
        }

        // my async copies done; barrier publishes the buffer to all waves
        if (kt + 1 < KT) wait_async_lds();
        __syncthreads();
    }

    // store C: VGPR v of a 16x16 f32 tile holds (M = v + 8*(lane>=16), N = lane&15)
#pragma unroll
    for (int i = 0; i < 4; ++i) {
#pragma unroll
        for (int j = 0; j < 4; ++j) {
            int row = m0 + wrow * 64 + i * 16 + ((lane >> 4) << 3);
            int col = n0 + wcol * 64 + j * 16 + laneM;
            float* p = out + (size_t)row * N_DIM + col;
#pragma unroll
            for (int v = 0; v < 8; ++v)
                p[(size_t)v * N_DIM] = acc[i][j][v];
        }
    }
}

extern "C" void kernel_launch(void* const* d_in, const int* in_sizes, int n_in,
                              void* d_out, int out_size, void* d_ws, size_t ws_size,
                              hipStream_t stream) {
    const float* x  = (const float*)d_in[0];   // [4,2048,4096]
    const float* W  = (const float*)d_in[1];   // [4096,4096]
    const float* lA = (const float*)d_in[2];   // [16,4096]
    const float* lB = (const float*)d_in[3];   // [4096,16]
    float* out = (float*)d_out;                // [4,2048,4096]

    unsigned short* xbf = (unsigned short*)d_ws;                 // 64 MB
    unsigned short* wbf = xbf + (size_t)M_DIM * K_DIM;           // +32 MB

    // x -> bf16 : 8192*4096/8 threads
    cvt_x_bf16<<<(M_DIM * (size_t)K_DIM) / 8 / 256, 256, 0, stream>>>(x, xbf);
    // W' = W + 2*B*A -> bf16 : 4096*4096/2 threads
    fuse_w_bf16<<<(N_DIM * (size_t)K_DIM) / 2 / 256, 256, 0, stream>>>(W, lA, lB, wbf);
    // GEMM
    dim3 grid(N_DIM / BN, M_DIM / BM);   // (32, 32)
    gemm_bf16_wmma<<<grid, 256, 0, stream>>>(xbf, wbf, out);
}
